// DLNN_13322988552585
// MI455X (gfx1250) — matrix operand order repaired and checked by reading, not compiled, not run on hardware
//
#include <hip/hip_runtime.h>

typedef __attribute__((ext_vector_type(16))) __bf16 v16bf;
typedef __attribute__((ext_vector_type(8)))  __bf16 v8bf;
typedef __attribute__((ext_vector_type(8)))  float  v8f;

#define BT_DIM (256*512)
#define I_DIM  512
#define L_DIM  128
#define ND_DIM 512   // L*D
#define T_DIM  512
#define B_DIM  256
#define O_DIM  2

__device__ __forceinline__ float fast_sigmoid(float v) {
    return __builtin_amdgcn_rcpf(1.f + __expf(-v));          // v_exp + v_rcp
}
__device__ __forceinline__ float fast_tanh(float y) {
    // tanh(y) = 1 - 2/(e^{2y}+1); saturates correctly at +-inf
    return 1.f - 2.f * __builtin_amdgcn_rcpf(__expf(2.f * y) + 1.f);
}

// ---------------- prep: W_in f32 -> bf16 (1 MB -> 512 KB, L2 resident) ------
__global__ __launch_bounds__(256) void prep_win_bf16(const float* __restrict__ Win,
                                                     __bf16* __restrict__ Wb) {
    int i = blockIdx.x * 256 + threadIdx.x;
    if (i < ND_DIM * I_DIM) Wb[i] = (__bf16)Win[i];
}

// ---------------- phase 1: merged_in[bt,l] = sum_d gate * relu(x @ W_in^T) --
// block = 256 thr = 8 waves = 2 M-tiles(16 rows) x 4 N-quarters(128 cols)
// x slab (32 rows x 512 K) staged once in LDS as bf16; A-fragments from LDS.
__global__ __launch_bounds__(256) void phase1_gemm(const float* __restrict__ x,
                                                   const __bf16* __restrict__ Wb,
                                                   const float* __restrict__ Wg,
                                                   float* __restrict__ merged) {
    __shared__ __bf16 s_x[32][I_DIM + 8];    // pad 8 -> row stride 260 dwords

    const int tid  = threadIdx.x;
    const int wave = tid >> 5;
    const int lane = tid & 31;
    const int half = lane >> 4;
    const int l16  = lane & 15;
    const int m0    = blockIdx.x * 32 + (wave >> 2) * 16;
    const int nbase = (wave & 3) * 128;

    // cooperative stage: 32 rows x 512 f32 -> bf16 LDS (coalesced float4)
    const float* xblk = x + (size_t)blockIdx.x * 32 * I_DIM;
    #pragma unroll
    for (int i = 0; i < 16; ++i) {
        int c    = tid + 256 * i;            // 4096 float4 chunks
        int row  = c >> 7;
        int col4 = (c & 127) * 4;
        float4 f = *(const float4*)(xblk + (size_t)row * I_DIM + col4);
        __bf16 h0 = (__bf16)f.x, h1 = (__bf16)f.y, h2 = (__bf16)f.z, h3 = (__bf16)f.w;
        s_x[row][col4 + 0] = h0; s_x[row][col4 + 1] = h1;
        s_x[row][col4 + 2] = h2; s_x[row][col4 + 3] = h3;
    }

    float gcol[8];
    #pragma unroll
    for (int t = 0; t < 8; ++t) gcol[t] = fast_sigmoid(Wg[nbase + t * 16 + l16]);

    v8f acc[8];
    #pragma unroll
    for (int t = 0; t < 8; ++t)
        #pragma unroll
        for (int r = 0; r < 8; ++r) acc[t][r] = 0.f;

    __syncthreads();
    const __bf16* arow = &s_x[(wave >> 2) * 16 + l16][0];

    #pragma unroll 2
    for (int kc = 0; kc < I_DIM; kc += 32) {
        // A fragment from LDS: lane<16 -> K in [kc..kc+15]; lane>=16 -> [kc+16..kc+31]
        const v8bf a0 = *(const v8bf*)(arow + kc + half * 16);
        const v8bf a1 = *(const v8bf*)(arow + kc + half * 16 + 8);
        v16bf a;
        #pragma unroll
        for (int e = 0; e < 8; ++e) { a[e] = a0[e]; a[8 + e] = a1[e]; }
        #pragma unroll
        for (int t = 0; t < 8; ++t) {
            const __bf16* pb = Wb + (size_t)(nbase + t * 16 + l16) * I_DIM + kc + half * 16;
            v16bf b;
            #pragma unroll
            for (int e = 0; e < 16; ++e) b[e] = pb[e];
            acc[t] = __builtin_amdgcn_wmma_f32_16x16x32_bf16(
                false, a, false, b, (short)0, acc[t], false, false);
        }
    }
    // epilogue: relu * gate, reduce groups of 4 N-columns (adjacent lanes)
    #pragma unroll
    for (int t = 0; t < 8; ++t) {
        #pragma unroll
        for (int r = 0; r < 8; ++r) {
            float v = fmaxf(acc[t][r], 0.f) * gcol[t];
            v += __shfl_xor(v, 1);
            v += __shfl_xor(v, 2);
            if ((lane & 3) == 0) {
                int n   = nbase + t * 16 + l16;
                int row = m0 + r + half * 8;
                merged[(size_t)row * L_DIM + (n >> 2)] = v;
            }
        }
    }
}

// ---------------- phase 2: serial recurrence, 16 batch rows per block -------
__global__ __launch_bounds__(256) void phase2_recur(
    const float* __restrict__ merged, const float* __restrict__ Wl,
    const float* __restrict__ Wg, const float* __restrict__ bl,
    const float* __restrict__ Wo, const float* __restrict__ bo,
    const float* __restrict__ lnw, const float* __restrict__ lnb,
    const float* __restrict__ leaky, float* __restrict__ out) {
    __shared__ float  s_state[16][L_DIM];
    __shared__ float  s_li[2][16][L_DIM];    // double-buffered merged_in slice (+rec)
    __shared__ __bf16 s_stbf[16][L_DIM + 8]; // pad -> stride 68 dwords, no conflicts

    const int tid  = threadIdx.x;
    const int wave = tid >> 5;
    const int lane = tid & 31;
    const int half = lane >> 4;
    const int l16  = lane & 15;
    const int b0   = blockIdx.x * 16;
    const float alpha  = fast_sigmoid(leaky[0]);
    const float oalpha = 1.f - alpha;

    for (int idx = tid; idx < 16 * L_DIM; idx += 256) (&s_state[0][0])[idx] = 0.f;
    for (int idx = tid; idx < 16 * (L_DIM + 8); idx += 256)
        (&s_stbf[0][0])[idx] = (__bf16)0.f;

    // W_liquid B-fragments: constant across all 512 steps -> keep in VGPRs
    v16bf Bf[4][4];
    #pragma unroll
    for (int t = 0; t < 4; ++t) {
        const int n = wave * 64 + t * 16 + l16;
        #pragma unroll
        for (int k = 0; k < 4; ++k) {
            const float* pb = Wl + (size_t)n * L_DIM + k * 32 + half * 16;
            #pragma unroll
            for (int e = 0; e < 16; ++e) Bf[t][k][e] = (__bf16)pb[e];
        }
    }
    float gcol[4];
    #pragma unroll
    for (int t = 0; t < 4; ++t) gcol[t] = fast_sigmoid(Wg[wave * 64 + t * 16 + l16]);

    // this lane's merged_in column (same for every row chunk) + its bias
    const int  lcol = wave * 16 + l16;       // wave owns l in [wave*16, wave*16+16)
    const float blc = bl[lcol];
    // preload slice for t=0
    #pragma unroll
    for (int it = 0; it < 8; ++it) {
        int row = it * 2 + half;
        s_li[0][row][lcol] = merged[(size_t)(b0 + row) * T_DIM * L_DIM + lcol] + blc;
    }
    __syncthreads();

    for (int ts = 0; ts < T_DIM; ++ts) {
        const int cur = ts & 1, nxt = cur ^ 1;
        // prefetch merged slice two steps ahead (global_prefetch_b8)
        if (ts + 2 < T_DIM) {
            int prow = lane & 15;
            const float* pf = &merged[((size_t)(b0 + prow) * T_DIM + ts + 2) * L_DIM
                                      + wave * 16 + (lane >> 4) * 8];
            __builtin_prefetch(pf, 0, 3);
        }
        // issue next-step slice loads early; consumed at end of this step
        float pre[8];
        if (ts + 1 < T_DIM) {
            #pragma unroll
            for (int it = 0; it < 8; ++it) {
                int row = it * 2 + half;
                pre[it] = merged[((size_t)(b0 + row) * T_DIM + ts + 1) * L_DIM + lcol];
            }
        }
        // recurrent GEMM: A = state (bf16 shadow in LDS), B = register frags
        v8f acc[4];
        #pragma unroll
        for (int t = 0; t < 4; ++t)
            #pragma unroll
            for (int r = 0; r < 8; ++r) acc[t][r] = 0.f;
        #pragma unroll
        for (int k = 0; k < 4; ++k) {
            const __bf16* ps = &s_stbf[l16][0];
            const int kb = k * 32;
            const v8bf a0 = *(const v8bf*)(ps + kb + half * 16);
            const v8bf a1 = *(const v8bf*)(ps + kb + half * 16 + 8);
            v16bf a;
            #pragma unroll
            for (int e = 0; e < 8; ++e) { a[e] = a0[e]; a[8 + e] = a1[e]; }
            #pragma unroll
            for (int t = 0; t < 4; ++t)
                acc[t] = __builtin_amdgcn_wmma_f32_16x16x32_bf16(
                    false, a, false, Bf[t][k], (short)0, acc[t], false, false);
        }
        // relu * gate, 4-lane reduce, accumulate into current li slice
        #pragma unroll
        for (int t = 0; t < 4; ++t) {
            #pragma unroll
            for (int r = 0; r < 8; ++r) {
                float v = fmaxf(acc[t][r], 0.f) * gcol[t];
                v += __shfl_xor(v, 1);
                v += __shfl_xor(v, 2);
                if ((lane & 3) == 0) {
                    int lc  = wave * 16 + t * 4 + (l16 >> 2);
                    int row = r + half * 8;
                    s_li[cur][row][lc] += v;
                }
            }
        }
        __syncthreads();
        // layernorm + tanh + leaky state update; wave w handles rows 2w, 2w+1
        #pragma unroll
        for (int rr = 0; rr < 2; ++rr) {
            int row = wave * 2 + rr;
            float vals[4];
            float s = 0.f, sq = 0.f;
            #pragma unroll
            for (int j = 0; j < 4; ++j) {
                float v = s_li[cur][row][lane + 32 * j];
                vals[j] = v; s += v; sq += v * v;
            }
            #pragma unroll
            for (int off = 16; off > 0; off >>= 1) {
                s  += __shfl_xor(s, off);
                sq += __shfl_xor(sq, off);
            }
            float mean = s * (1.f / 128.f);
            float var  = sq * (1.f / 128.f) - mean * mean;
            float rstd = __builtin_amdgcn_rsqf(var + 1e-5f);
            #pragma unroll
            for (int j = 0; j < 4; ++j) {
                int lc  = lane + 32 * j;
                float y  = (vals[j] - mean) * rstd * lnw[lc] + lnb[lc];
                float ns = oalpha * s_state[row][lc] + alpha * fast_tanh(y);
                s_state[row][lc] = ns;
                s_stbf[row][lc]  = (__bf16)ns;
            }
        }
        // park prefetched next slice (+bias) into the other li buffer
        if (ts + 1 < T_DIM) {
            #pragma unroll
            for (int it = 0; it < 8; ++it) {
                int row = it * 2 + half;
                s_li[nxt][row][lcol] = pre[it] + blc;
            }
        }
        __syncthreads();
    }
    // output projection: [16,128] @ W_out^T + b_out  (O=2)
    if (tid < 32) {
        int b = tid >> 1, o = tid & 1;
        float accv = bo[o];
        for (int l = 0; l < L_DIM; ++l) accv += s_state[b][l] * Wo[o * L_DIM + l];
        out[(size_t)(b0 + b) * O_DIM + o] = accv;
    }
}

extern "C" void kernel_launch(void* const* d_in, const int* in_sizes, int n_in,
                              void* d_out, int out_size, void* d_ws, size_t ws_size,
                              hipStream_t stream) {
    const float* x     = (const float*)d_in[0];
    const float* Win   = (const float*)d_in[1];
    const float* Wl    = (const float*)d_in[2];
    const float* Wg    = (const float*)d_in[3];
    const float* bl    = (const float*)d_in[4];
    const float* Wo    = (const float*)d_in[5];
    const float* bo    = (const float*)d_in[6];
    const float* lnw   = (const float*)d_in[7];
    const float* lnb   = (const float*)d_in[8];
    const float* leaky = (const float*)d_in[9];
    float* out = (float*)d_out;

    float*  merged = (float*)d_ws;                                   // 67.1 MB
    __bf16* Wb     = (__bf16*)((char*)d_ws + (size_t)BT_DIM * L_DIM * sizeof(float));

    prep_win_bf16<<<(ND_DIM * I_DIM + 255) / 256, 256, 0, stream>>>(Win, Wb);
    phase1_gemm<<<BT_DIM / 32, 256, 0, stream>>>(x, Wb, Wg, merged);
    phase2_recur<<<B_DIM / 16, 256, 0, stream>>>(merged, Wl, Wg, bl, Wo, bo,
                                                 lnw, lnb, leaky, out);
}
